// MessageLayer_73134703116342
// MI455X (gfx1250) — matrix operand order repaired and checked by reading, not compiled, not run on hardware
//
#include <hip/hip_runtime.h>
#include <hip/hip_bf16.h>

typedef __attribute__((ext_vector_type(16))) _Float16 v16h;
typedef __attribute__((ext_vector_type(8)))  _Float16 v8h;
typedef __attribute__((ext_vector_type(8)))  float    v8f;
typedef __attribute__((ext_vector_type(4)))  float    v4f;

#define NN   25000      // nodes
#define NE   800000     // edges
#define DF   64         // feature dim
#define HID  256        // hidden
#define NH   3          // heads
#define TWO_D 128
#define NEG_SLOPE 0.01f

extern __shared__ char dynsm[];   // dynamic LDS pool (partitioned per kernel)

__device__ __forceinline__ float leaky(float x) { return x >= 0.f ? x : NEG_SLOPE * x; }

// order-preserving float -> uint key (for atomic segment-max)
__device__ __forceinline__ unsigned fkey(float f) {
  unsigned u = __float_as_uint(f);
  return (u & 0x80000000u) ? ~u : (u | 0x80000000u);
}
__device__ __forceinline__ float fkey_dec(unsigned k) {
  unsigned u = (k & 0x80000000u) ? (k ^ 0x80000000u) : ~k;
  return __uint_as_float(u);
}

__device__ __forceinline__ v8f wmma_f16(v16h a, v16h b, v8f c) {
  return __builtin_amdgcn_wmma_f32_16x16x32_f16(false, a, false, b, (short)0, c, false, false);
}

// CDNA5 async global->LDS copy (ASYNCcnt tracked), ISA §15.18.3 opcode 98.
// ldsAddr: LDS byte address (low 32 bits of generic shared pointer = LDS offset).
__device__ __forceinline__ void async_ld_b128(unsigned ldsAddr, const void* gptr) {
  asm volatile("global_load_async_to_lds_b128 %0, %1, off"
               :: "v"(ldsAddr), "v"(gptr) : "memory");
}
__device__ __forceinline__ void wait_async0() {
  asm volatile("s_wait_asynccnt 0x0" ::: "memory");
}

// Build one 16x32 f16 A-fragment (K-chunk base kc) for this lane.
// Lanes 0-15 halves: K = kc+[0..7], kc+16+[0..7]; lanes 16-31: +8 / +24.
__device__ __forceinline__ v16h gather_A(const float* __restrict__ fs,
                                         const float* __restrict__ fn,
                                         int kc, int grp) {
  int k0 = kc + 8 * grp;
  int k1 = kc + 16 + 8 * grp;
  const float* p0 = (k0 < DF) ? (fs + k0) : (fn + (k0 - DF));
  const float* p1 = (k1 < DF) ? (fs + k1) : (fn + (k1 - DF));
  v4f x0 = *(const v4f*)(p0);
  v4f x1 = *(const v4f*)(p0 + 4);
  v4f y0 = *(const v4f*)(p1);
  v4f y1 = *(const v4f*)(p1 + 4);
  v16h a;
#pragma unroll
  for (int i = 0; i < 4; ++i) {
    a[i]      = (_Float16)x0[i];
    a[i + 4]  = (_Float16)x1[i];
    a[i + 8]  = (_Float16)y0[i];
    a[i + 12] = (_Float16)y1[i];
  }
  return a;
}

// ---------------- weight prepack: f32 [H][K][N] -> f16 B fragments [H][NT][KC][32][16]
__global__ void pack_b_kernel(const float* __restrict__ src, _Float16* __restrict__ dst,
                              int K, int N) {
  int KC = K >> 5, NT = N >> 4;
  int per = KC * NT * 512;
  int idx = blockIdx.x * blockDim.x + threadIdx.x;
  if (idx >= NH * per) return;
  int h = idx / per, rem = idx % per;
  int t = rem / (KC * 512);
  int rem2 = rem % (KC * 512);
  int c = rem2 / 512;
  int q = rem2 % 512;
  int L = q >> 4, i = q & 15;
  int kk = c * 32 + (i < 8 ? 8 * (L >> 4) + i : 16 + 8 * (L >> 4) + (i - 8));   // B: lanes 0-15 hold K 0..7,16..23; lanes 16-31 hold K 8..15,24..31
  int n = t * 16 + (L & 15);
  dst[idx] = (_Float16)src[(size_t)h * K * N + (size_t)kk * N + n];
}

// ---------------- init: residual + clear segment buffers
__global__ void init_kernel(const float* __restrict__ prev, float* __restrict__ out,
                            unsigned* __restrict__ gmaxKey, float* __restrict__ denom) {
  int i = blockIdx.x * blockDim.x + threadIdx.x;
  if (i < NN * DF) out[i] = prev[i];
  if (i < NH * NN) { gmaxKey[i] = 0u; denom[i] = 0.f; }
}

// ---------------- pass 1: gate logits per (head, edge) + atomic segment max
// 8 waves/block, 16 edges/wave; per-head gate_W1 fragments staged in LDS (64 KB) via async copy.
__global__ __launch_bounds__(256) void gate_kernel(
    const float* __restrict__ prev, const int* __restrict__ self_idx,
    const int* __restrict__ nbr_idx, const _Float16* __restrict__ gatePack,
    const float* __restrict__ gb1, const float* __restrict__ gW2,
    const float* __restrict__ gb2, float* __restrict__ gateVal,
    unsigned* __restrict__ gmaxKey) {
  _Float16* smB = (_Float16*)dynsm;                       // 32768 halfs = 64 KB
  unsigned smBase = (unsigned)(uintptr_t)smB;
  int wave = threadIdx.x >> 5;
  int lane = threadIdx.x & 31;
  int tile = blockIdx.x * 8 + wave;
  if (tile * 16 >= NE) return;               // grid exact; wave-uniform
  int base = tile * 16;
  int m = lane & 15, grp = lane >> 4;
  int e = base + m;
  const float* fs = prev + (size_t)self_idx[e] * DF;
  const float* fn = prev + (size_t)nbr_idx[e] * DF;
  v16h a[4];
#pragma unroll
  for (int c = 0; c < 4; ++c) a[c] = gather_A(fs, fn, c * 32, grp);

  for (int h = 0; h < NH; ++h) {
    // stage this head's 64 KB of B fragments: 4096 x 16B async chunks, 16 per thread
    const _Float16* wp = gatePack + (size_t)h * (16 * 4 * 512);
#pragma unroll
    for (int j = 0; j < 16; ++j) {
      int idx = j * 256 + threadIdx.x;
      async_ld_b128(smBase + idx * 16, wp + idx * 8);
    }
    wait_async0();
    __syncthreads();

    float acc[8];
#pragma unroll
    for (int r = 0; r < 8; ++r) acc[r] = 0.f;
#pragma unroll 4
    for (int t = 0; t < 16; ++t) {
      v8f C = {0.f, 0.f, 0.f, 0.f, 0.f, 0.f, 0.f, 0.f};
#pragma unroll
      for (int c = 0; c < 4; ++c) {
        v16h b = *(const v16h*)(smB + (((t * 4 + c) * 32 + lane) << 4));
        C = wmma_f16(a[c], b, C);
      }
      float bj = gb1[h * HID + t * 16 + m];
      float w2 = gW2[h * HID + t * 16 + m];
#pragma unroll
      for (int r = 0; r < 8; ++r) acc[r] += leaky(C[r] + bj) * w2;
    }
    // reduce across the 16 columns (lanes sharing lane>>4)
#pragma unroll
    for (int off = 1; off < 16; off <<= 1)
#pragma unroll
      for (int r = 0; r < 8; ++r) acc[r] += __shfl_xor(acc[r], off, 32);
    if (m == 0) {
      float b2 = gb2[h];
#pragma unroll
      for (int r = 0; r < 8; ++r) {
        int ee = base + r + 8 * grp;
        float g = acc[r] + b2;
        gateVal[h * NE + ee] = g;
        atomicMax(&gmaxKey[h * NN + self_idx[ee]], fkey(g));
      }
    }
    __syncthreads();   // all waves done with smB before next head restages
  }
}

// ---------------- pass 2: e = w^p * exp(g - gmax); denom += e
__global__ void denom_kernel(const int* __restrict__ self_idx, const int* __restrict__ nbr_idx,
                             const float* __restrict__ node_w, const float* __restrict__ pow_p,
                             const unsigned* __restrict__ gmaxKey,
                             float* __restrict__ gateVal, float* __restrict__ denom) {
  int i = blockIdx.x * blockDim.x + threadIdx.x;
  if (i >= NH * NE) return;
  int h = i / NE;
  int e = i - h * NE;
  int s = self_idx[e];
  float g = gateVal[i];
  float gmax = fkey_dec(gmaxKey[h * NN + s]);
  float w = node_w[nbr_idx[e]];
  float ev = powf(w, pow_p[h]) * expf(g - gmax);
  gateVal[i] = ev;
  atomicAdd(&denom[h * NN + s], ev);
}

// ---------------- pass 3: message MLP (recomputed), weighted head-mean, scatter-add
// 4 waves/block. LDS: W1 frags 64 KB + W2 frags 32 KB + h1 staging 32 KB = 128 KB.
#define WAVES3 4
__global__ __launch_bounds__(128) void message_kernel(
    const float* __restrict__ prev, const int* __restrict__ self_idx,
    const int* __restrict__ nbr_idx, const _Float16* __restrict__ msg1Pack,
    const _Float16* __restrict__ msg2Pack, const float* __restrict__ mb1,
    const float* __restrict__ mb2, const float* __restrict__ gateVal,
    const float* __restrict__ denom, float* __restrict__ out) {
  _Float16* smW1 = (_Float16*)dynsm;          // 32768 halfs
  _Float16* smW2 = smW1 + 32768;              // 16384 halfs
  _Float16* smH1 = smW2 + 16384;              // WAVES3 * 4096 halfs
  unsigned w1Base = (unsigned)(uintptr_t)smW1;
  unsigned w2Base = (unsigned)(uintptr_t)smW2;

  int wave = threadIdx.x >> 5;
  int lane = threadIdx.x & 31;
  int tile = blockIdx.x * WAVES3 + wave;
  if (tile * 16 >= NE) return;               // grid exact; wave-uniform
  int base = tile * 16;
  int m = lane & 15, grp = lane >> 4;
  int e = base + m;
  const float* fs = prev + (size_t)self_idx[e] * DF;
  const float* fn = prev + (size_t)nbr_idx[e] * DF;
  v16h a[4];
#pragma unroll
  for (int c = 0; c < 4; ++c) a[c] = gather_A(fs, fn, c * 32, grp);

  const v8f zero8 = {0.f, 0.f, 0.f, 0.f, 0.f, 0.f, 0.f, 0.f};
  v8f outacc[4];
#pragma unroll
  for (int t2 = 0; t2 < 4; ++t2) outacc[t2] = zero8;

  _Float16* myH = smH1 + wave * (16 * HID);

  for (int h = 0; h < NH; ++h) {
    // stage W1 (4096 chunks) + W2 (2048 chunks) via async copy, 128 threads
    const _Float16* w1p = msg1Pack + (size_t)h * (16 * 4 * 512);
    const _Float16* w2p = msg2Pack + (size_t)h * (4 * 8 * 512);
#pragma unroll
    for (int j = 0; j < 32; ++j) {
      int idx = j * 128 + threadIdx.x;
      async_ld_b128(w1Base + idx * 16, w1p + idx * 8);
    }
#pragma unroll
    for (int j = 0; j < 16; ++j) {
      int idx = j * 128 + threadIdx.x;
      async_ld_b128(w2Base + idx * 16, w2p + idx * 8);
    }
    wait_async0();
    __syncthreads();

    // normalized gate coefficients for my 8 C-rows (edges base + r + 8*grp)
    float cf[8];
#pragma unroll
    for (int r = 0; r < 8; ++r) {
      int ee = base + r + 8 * grp;
      int s = self_idx[ee];
      cf[r] = gateVal[h * NE + ee] / (denom[h * NN + s] + 1e-10f) * (1.f / (float)NH);
    }
    // GEMM1: [16x128]x[128x256], leaky, stage to LDS as f16
#pragma unroll 4
    for (int t = 0; t < 16; ++t) {
      v8f C = zero8;
#pragma unroll
      for (int c = 0; c < 4; ++c) {
        v16h b = *(const v16h*)(smW1 + (((t * 4 + c) * 32 + lane) << 4));
        C = wmma_f16(a[c], b, C);
      }
      float bj = mb1[h * HID + t * 16 + m];
#pragma unroll
      for (int r = 0; r < 8; ++r)
        myH[(r + 8 * grp) * HID + t * 16 + m] = (_Float16)leaky(C[r] + bj);
    }
    __syncthreads();
    // GEMM2: [16x256]x[256x64]; A re-read from LDS in A-fragment layout
    v8f macc[4];
#pragma unroll
    for (int t2 = 0; t2 < 4; ++t2) macc[t2] = zero8;
#pragma unroll 2
    for (int c2 = 0; c2 < 8; ++c2) {
      int k0 = c2 * 32 + 8 * grp;
      int k1 = c2 * 32 + 16 + 8 * grp;
      v8h lo = *(const v8h*)(myH + m * HID + k0);
      v8h hi = *(const v8h*)(myH + m * HID + k1);
      v16h a2;
#pragma unroll
      for (int i = 0; i < 8; ++i) { a2[i] = lo[i]; a2[i + 8] = hi[i]; }
#pragma unroll
      for (int t2 = 0; t2 < 4; ++t2) {
        v16h b = *(const v16h*)(smW2 + (((t2 * 8 + c2) * 32 + lane) << 4));
        macc[t2] = wmma_f16(a2, b, macc[t2]);
      }
    }
#pragma unroll
    for (int t2 = 0; t2 < 4; ++t2) {
      float b2 = mb2[h * DF + t2 * 16 + m];
#pragma unroll
      for (int r = 0; r < 8; ++r) outacc[t2][r] += cf[r] * (macc[t2][r] + b2);
    }
    __syncthreads();   // LDS (weights + h1) reusable for next head
  }
  // scatter-add into out (out already holds the residual prev)
#pragma unroll
  for (int r = 0; r < 8; ++r) {
    int s = self_idx[base + r + 8 * grp];
#pragma unroll
    for (int t2 = 0; t2 < 4; ++t2)
      atomicAdd(out + (size_t)s * DF + t2 * 16 + m, outacc[t2][r]);
  }
}

static inline size_t align256(size_t x) { return (x + 255) & ~(size_t)255; }

extern "C" void kernel_launch(void* const* d_in, const int* in_sizes, int n_in,
                              void* d_out, int out_size, void* d_ws, size_t ws_size,
                              hipStream_t stream) {
  const float* node_w   = (const float*)d_in[0];
  const float* prev     = (const float*)d_in[1];
  const int*   self_idx = (const int*)d_in[2];
  const int*   nbr_idx  = (const int*)d_in[3];
  const float* gW1 = (const float*)d_in[4];
  const float* gb1 = (const float*)d_in[5];
  const float* gW2 = (const float*)d_in[6];
  const float* gb2 = (const float*)d_in[7];
  const float* mW1 = (const float*)d_in[8];
  const float* mb1 = (const float*)d_in[9];
  const float* mW2 = (const float*)d_in[10];
  const float* mb2 = (const float*)d_in[11];
  const float* pow_p = (const float*)d_in[12];
  float* out = (float*)d_out;

  // workspace layout (~10.7 MB)
  char* ws = (char*)d_ws;
  size_t off = 0;
  _Float16* gatePack = (_Float16*)(ws + off); off = align256(off + (size_t)NH * 32768 * 2);
  _Float16* msg1Pack = (_Float16*)(ws + off); off = align256(off + (size_t)NH * 32768 * 2);
  _Float16* msg2Pack = (_Float16*)(ws + off); off = align256(off + (size_t)NH * 16384 * 2);
  float*    gateVal  = (float*)(ws + off);    off = align256(off + (size_t)NH * NE * 4);
  unsigned* gmaxKey  = (unsigned*)(ws + off); off = align256(off + (size_t)NH * NN * 4);
  float*    denom    = (float*)(ws + off);    off = align256(off + (size_t)NH * NN * 4);
  (void)ws_size;

  // prepack weights to f16 B fragments
  pack_b_kernel<<<(NH * 32768 + 255) / 256, 256, 0, stream>>>(gW1, gatePack, TWO_D, HID);
  pack_b_kernel<<<(NH * 32768 + 255) / 256, 256, 0, stream>>>(mW1, msg1Pack, TWO_D, HID);
  pack_b_kernel<<<(NH * 16384 + 255) / 256, 256, 0, stream>>>(mW2, msg2Pack, HID, DF);

  // residual + clear segment buffers
  init_kernel<<<(NN * DF + 255) / 256, 256, 0, stream>>>(prev, out, gmaxKey, denom);

  // pass 1: gate logits + segment max  (8 waves/block, 16 edges/wave, 64 KB dyn LDS)
  int tiles = NE / 16;                           // 50000
  gate_kernel<<<tiles / 8, 256, 65536, stream>>>(prev, self_idx, nbr_idx, gatePack,
                                                 gb1, gW2, gb2, gateVal, gmaxKey);

  // pass 2: exp / denominator
  denom_kernel<<<(NH * NE + 255) / 256, 256, 0, stream>>>(self_idx, nbr_idx, node_w, pow_p,
                                                          gmaxKey, gateVal, denom);

  // pass 3: message GEMMs + weighted scatter (4 waves/block, 128 KB dyn LDS)
  message_kernel<<<tiles / WAVES3, 32 * WAVES3, 131072, stream>>>(prev, self_idx, nbr_idx,
                                                                  msg1Pack, msg2Pack, mb1, mb2,
                                                                  gateVal, denom, out);
}
